// ProbSparseAttention_618475291269
// MI455X (gfx1250) — matrix operand order repaired and checked by reading, not compiled, not run on hardware
//
#include <hip/hip_runtime.h>
#include <hip/hip_bf16.h>
#include <math.h>

// Problem constants from the reference (B=8, L=4096, d=512).
#define BATCH 8
#define SEQL  4096
#define DIM   512
#define KTOP  41                 // int(5.0 * ln(4096 + 1e-9)) = 41
#define SCALE 0.044194173824159216f  // 1/sqrt(512)

#define LDS_PAD   8              // halves; row stride 520 halves = 260 dwords == 4 (mod 64)
#define ROW_HALVES (DIM + LDS_PAD)

typedef __attribute__((ext_vector_type(16))) __bf16 v16bf;
typedef __attribute__((ext_vector_type(8)))  float  v8f;

union Frag16 {
  v16bf v;
  unsigned short u[16];
  uint4 q[2];
};

__device__ __forceinline__ unsigned short f2bf(float f) {
  union { float f; unsigned u; } cv; cv.f = f;
  unsigned u = cv.u;
  u += 0x7FFFu + ((u >> 16) & 1u);   // round-to-nearest-even
  return (unsigned short)(u >> 16);
}

// CDNA5 async global->LDS copy (VGLOBAL encoding, tracked by ASYNCcnt).
// VDST VGPR = per-lane LDS byte offset; VADDR = 64-bit global address.
__device__ __forceinline__ void async_load_b128_to_lds(
    unsigned lds_off, const void* gaddr) {
  asm volatile("global_load_async_to_lds_b128 %0, %1, off"
               :: "v"(lds_off), "v"(gaddr)
               : "memory");
}
__device__ __forceinline__ void wait_asynccnt_le4() {
  asm volatile("s_wait_asynccnt 0x4" ::: "memory");
}
__device__ __forceinline__ void wait_asynccnt_zero() {
  asm volatile("s_wait_asynccnt 0x0" ::: "memory");
}

// ---------------------------------------------------------------------------
// 1) K (fp32) -> Kh (bf16 raw bits) elementwise
// ---------------------------------------------------------------------------
__global__ __launch_bounds__(256) void convert_bf16_kernel(
    const float* __restrict__ src, unsigned short* __restrict__ dst, int n) {
  int i = blockIdx.x * 256 + threadIdx.x;
  if (i < n) dst[i] = f2bf(src[i]);
}

// ---------------------------------------------------------------------------
// 2) Column means over L for K and V: out[b, d] = mean_l X[b, l, d]
// ---------------------------------------------------------------------------
__global__ __launch_bounds__(256) void means_kernel(
    const float* __restrict__ K, const float* __restrict__ V,
    float* __restrict__ kmean, float* __restrict__ vmean) {
  int b     = blockIdx.x >> 1;
  int which = blockIdx.x & 1;
  const float* src = which ? V : K;
  float*       dst = which ? vmean : kmean;
  int d0 = threadIdx.x;           // two columns per thread: d0 and d0+256
  float s0 = 0.f, s1 = 0.f;
  const float* base = src + (size_t)b * SEQL * DIM;
  for (int l = 0; l < SEQL; ++l) {
    const float* row = base + (size_t)l * DIM;
    s0 += row[d0];
    s1 += row[d0 + 256];
  }
  dst[(size_t)b * DIM + d0]       = s0 * (1.0f / SEQL);
  dst[(size_t)b * DIM + d0 + 256] = s1 * (1.0f / SEQL);
}

// ---------------------------------------------------------------------------
// 3) Dense score rowmax via WMMA bf16 with async-to-LDS K streaming.
//    Grid: BATCH * (SEQL/128) blocks of 256 threads (8 waves).
//    Each wave owns 16 query rows (A fragments register-resident for all of
//    d=512). K is streamed once per block in 16-key x 512-d tiles via
//    GLOBAL_LOAD_ASYNC_TO_LDS_B128, triple-buffered (depth-2 pipeline, no
//    VGPR staging, one barrier per tile), shared by all 8 waves.
//    Per tile: two independent WMMA accumulator chains (even/odd d-chunks)
//    so ds_load latency overlaps matrix execution.
// ---------------------------------------------------------------------------
__global__ __launch_bounds__(256) void rowmax_wmma_kernel(
    const float* __restrict__ Q, const unsigned short* __restrict__ Kh,
    float* __restrict__ rowmax_out) {
  __shared__ __align__(16) unsigned short kbuf[3][16][ROW_HALVES];

  const int tid    = threadIdx.x;
  const int b      = blockIdx.x >> 5;        // 32 blocks per batch
  const int qchunk = blockIdx.x & 31;        // 128 queries per block
  const int wave   = tid >> 5;
  const int lane   = tid & 31;
  const int row    = lane & 15;
  const int hi     = lane >> 4;              // lane group 0/1
  const int q0     = qchunk * 128 + wave * 16;

  // ---- A fragments: Q rows q0..q0+15, d=512 as 16 chunks of 32 (bf16).
  // 16-bit A 16x32, contiguous k order: lanes 0-15 hold K 0..15 of row=lane,
  // lanes 16-31 hold K 16..31 of row=lane-16.
  const float* qrow = Q + ((size_t)b * SEQL + (q0 + row)) * DIM;
  Frag16 A[16];
#pragma unroll
  for (int kc = 0; kc < 16; ++kc) {
    const float* p0 = qrow + kc * 32 + hi * 16;  // elems 0..7
    const float* p1 = p0 + 8;                    // elems 8..15
#pragma unroll
    for (int t = 0; t < 8; ++t) A[kc].u[t]     = f2bf(p0[t]);
#pragma unroll
    for (int t = 0; t < 8; ++t) A[kc].u[8 + t] = f2bf(p1[t]);
  }

  float rmax[8];
#pragma unroll
  for (int r = 0; r < 8; ++r) rmax[r] = -INFINITY;

  const unsigned short* kbase = Kh + (size_t)b * SEQL * DIM;

  // Cooperative tile fetch: 16 threads per key row, 64B (4 x b128) each.
  const int frow = tid >> 4;                 // 0..15 key row within tile
  const int fseg = (tid & 15) * 32;          // half offset within row (64B)

  // Depth-2 async pipeline prologue: tiles 0 and 1 in flight.
#pragma unroll
  for (int pre = 0; pre < 2; ++pre) {
    const unsigned short* gp = kbase + (size_t)(pre * 16 + frow) * DIM + fseg;
#pragma unroll
    for (int i = 0; i < 4; ++i) {
      unsigned loff = (unsigned)(size_t)(&kbuf[pre][frow][fseg + i * 8]);
      async_load_b128_to_lds(loff, gp + i * 8);
    }
  }

  const int NT = SEQL / 16;                  // 256 tiles
  for (int t = 0; t < NT; ++t) {
    // Retire tile t (4 loads of tile t+1 may stay in flight; on the final
    // iteration nothing else is outstanding so wait for zero).
    if (t + 1 < NT) wait_asynccnt_le4();
    else            wait_asynccnt_zero();
    // Single barrier: (a) publishes tile t to all waves, (b) proves every
    // wave finished compute t-1, i.e. the buffer tile t+2 will overwrite
    // (last read at t-1) is free.
    __syncthreads();

    if (t + 2 < NT) {
      const unsigned short* gp =
          kbase + (size_t)((t + 2) * 16 + frow) * DIM + fseg;
#pragma unroll
      for (int i = 0; i < 4; ++i) {
        unsigned loff =
            (unsigned)(size_t)(&kbuf[(t + 2) % 3][frow][fseg + i * 8]);
        async_load_b128_to_lds(loff, gp + i * 8);
      }
    }

    // 16 x v_wmma_f32_16x16x32_bf16 on the resident tile, as two
    // independent accumulator chains (even / odd d-chunks) so the
    // scheduler can overlap ds_load_b128 latency with matrix execution.
    // 16-bit B 32x16, same contiguous k order as A: lanes 0-15 hold K 0..15 of col=lane,
    // lanes 16-31 hold K 16..31 of col=lane-16 (contiguous 16 halves).
    const unsigned short* bufrow = &kbuf[t % 3][row][hi * 16];
    v8f acc0 = {}, acc1 = {};
#pragma unroll
    for (int kc = 0; kc < 16; kc += 2) {
      Frag16 B0, B1;
      const uint4* bp0 = (const uint4*)(bufrow + kc * 32);
      const uint4* bp1 = (const uint4*)(bufrow + (kc + 1) * 32);
      B0.q[0] = bp0[0]; B0.q[1] = bp0[1];
      B1.q[0] = bp1[0]; B1.q[1] = bp1[1];
      acc0 = __builtin_amdgcn_wmma_f32_16x16x32_bf16(
          false, A[kc].v, false, B0.v, (short)0, acc0, false, false);
      acc1 = __builtin_amdgcn_wmma_f32_16x16x32_bf16(
          false, A[kc + 1].v, false, B1.v, (short)0, acc1, false, false);
    }
#pragma unroll
    for (int r = 0; r < 8; ++r)
      rmax[r] = fmaxf(rmax[r], acc0[r] + acc1[r]);
  }

  // C layout: VGPR r holds row (r + hi*8), N = lane&15. Reduce across the
  // 16-lane half (xor masks 1..8 never cross the half boundary).
#pragma unroll
  for (int r = 0; r < 8; ++r) {
    float m = rmax[r];
#pragma unroll
    for (int ofs = 8; ofs >= 1; ofs >>= 1)
      m = fmaxf(m, __shfl_xor(m, ofs, 32));
    if ((lane & 15) == 0)
      rowmax_out[(size_t)b * SEQL + q0 + hi * 8 + r] = m;
  }
}

// ---------------------------------------------------------------------------
// 4) M[b,q] = scale * (rowmax[b,q] - Q[b,q,:] . K_mean[b,:])
//    (rowmean(Q K^T) == Q . mean(K), so no second GEMM needed)
// ---------------------------------------------------------------------------
__global__ __launch_bounds__(128) void compute_m_kernel(
    const float* __restrict__ Q, const float* __restrict__ kmean,
    const float* __restrict__ rowmax, float* __restrict__ M) {
  const int wave = threadIdx.x >> 5;
  const int lane = threadIdx.x & 31;
  const int qid  = blockIdx.x * 4 + wave;     // 0 .. B*L-1
  const int b    = qid >> 12;
  const float* qp = Q + (size_t)qid * DIM;
  const float* km = kmean + (size_t)b * DIM;
  float s = 0.f;
#pragma unroll
  for (int i = 0; i < DIM / 32; ++i)
    s += qp[lane + 32 * i] * km[lane + 32 * i];
#pragma unroll
  for (int ofs = 16; ofs >= 1; ofs >>= 1) s += __shfl_xor(s, ofs, 32);
  if (lane == 0) M[qid] = (rowmax[qid] - s) * SCALE;
}

// ---------------------------------------------------------------------------
// 5) Top-KTOP per batch by iterated block argmax (ties -> smaller index).
// ---------------------------------------------------------------------------
__global__ __launch_bounds__(256) void topk_kernel(
    const float* __restrict__ M, int* __restrict__ topk) {
  __shared__ float sv[SEQL];
  __shared__ float rv[256];
  __shared__ int   ri[256];
  const int b   = blockIdx.x;
  const int tid = threadIdx.x;
  for (int j = tid; j < SEQL; j += 256) sv[j] = M[(size_t)b * SEQL + j];
  __syncthreads();
  for (int it = 0; it < KTOP; ++it) {
    float bv = -INFINITY; int bi = SEQL;
    for (int j = tid; j < SEQL; j += 256) {
      float x = sv[j];
      if (x > bv) { bv = x; bi = j; }
    }
    rv[tid] = bv; ri[tid] = bi;
    __syncthreads();
    for (int s = 128; s > 0; s >>= 1) {
      if (tid < s) {
        if (rv[tid + s] > rv[tid] ||
            (rv[tid + s] == rv[tid] && ri[tid + s] < ri[tid])) {
          rv[tid] = rv[tid + s];
          ri[tid] = ri[tid + s];
        }
      }
      __syncthreads();
    }
    if (tid == 0) {
      topk[b * KTOP + it] = ri[0];
      sv[ri[0]] = -INFINITY;
    }
    __syncthreads();
  }
}

// ---------------------------------------------------------------------------
// 6) out[b, l, d] = V_mean[b, d]
// ---------------------------------------------------------------------------
__global__ __launch_bounds__(256) void fill_mean_kernel(
    const float* __restrict__ vmean, float* __restrict__ out) {
  const size_t i = (size_t)blockIdx.x * 256 + threadIdx.x;
  const int b = (int)(i >> 21);          // L*DIM = 2^21
  const int d = (int)(i & (DIM - 1));
  out[i] = vmean[(size_t)b * DIM + d];
}

// ---------------------------------------------------------------------------
// 7) Sparse attention for the selected queries (tiny FLOP count):
//    grid = (KTOP, BATCH), 256 threads. Two-pass softmax, LDS score row.
// ---------------------------------------------------------------------------
__global__ __launch_bounds__(256) void sparse_attn_kernel(
    const float* __restrict__ Q, const float* __restrict__ K,
    const float* __restrict__ V, const int* __restrict__ topk,
    float* __restrict__ out) {
  __shared__ float qv[DIM];
  __shared__ float sc[SEQL];
  __shared__ float red[256];
  const int tid = threadIdx.x;
  const int b   = blockIdx.y;
  const int q   = topk[b * KTOP + blockIdx.x];

  const float* qp = Q + ((size_t)b * SEQL + q) * DIM;
  qv[tid]       = qp[tid];
  qv[tid + 256] = qp[tid + 256];
  __syncthreads();

  // scores + local max
  const float* kb = K + (size_t)b * SEQL * DIM;
  float lmax = -INFINITY;
  for (int j = tid; j < SEQL; j += 256) {
    const float4* kr = (const float4*)(kb + (size_t)j * DIM);
    float s = 0.f;
#pragma unroll 4
    for (int dd = 0; dd < DIM / 4; ++dd) {
      float4 kv = kr[dd];
      const float4* qq = (const float4*)&qv[dd * 4];
      float4 qvv = *qq;
      s += qvv.x * kv.x + qvv.y * kv.y + qvv.z * kv.z + qvv.w * kv.w;
    }
    s *= SCALE;
    sc[j] = s;
    lmax = fmaxf(lmax, s);
  }
  red[tid] = lmax;
  __syncthreads();
  for (int s = 128; s > 0; s >>= 1) {
    if (tid < s) red[tid] = fmaxf(red[tid], red[tid + s]);
    __syncthreads();
  }
  const float smax = red[0];
  __syncthreads();

  // exp + sum
  float lsum = 0.f;
  for (int j = tid; j < SEQL; j += 256) {
    float p = __expf(sc[j] - smax);
    sc[j] = p;
    lsum += p;
  }
  red[tid] = lsum;
  __syncthreads();
  for (int s = 128; s > 0; s >>= 1) {
    if (tid < s) red[tid] += red[tid + s];
    __syncthreads();
  }
  const float inv_sum = 1.0f / red[0];
  __syncthreads();

  // context[d] = sum_j p[j] * V[b, j, d]
  const float* vb = V + (size_t)b * SEQL * DIM;
  float* op = out + ((size_t)b * SEQL + q) * DIM;
#pragma unroll
  for (int half = 0; half < 2; ++half) {
    const int d = tid + half * 256;
    float acc = 0.f;
    for (int j = 0; j < SEQL; j += 4) {
      acc += sc[j + 0] * vb[(size_t)(j + 0) * DIM + d];
      acc += sc[j + 1] * vb[(size_t)(j + 1) * DIM + d];
      acc += sc[j + 2] * vb[(size_t)(j + 2) * DIM + d];
      acc += sc[j + 3] * vb[(size_t)(j + 3) * DIM + d];
    }
    op[d] = acc * inv_sum;
  }
}

// ---------------------------------------------------------------------------
extern "C" void kernel_launch(void* const* d_in, const int* in_sizes, int n_in,
                              void* d_out, int out_size, void* d_ws, size_t ws_size,
                              hipStream_t stream) {
  const float* Q = (const float*)d_in[0];
  const float* K = (const float*)d_in[1];
  const float* V = (const float*)d_in[2];
  float* out = (float*)d_out;

  const size_t n_elems = (size_t)BATCH * SEQL * DIM;   // 16,777,216

  // Workspace layout (~34 MB)
  char* ws = (char*)d_ws;
  unsigned short* Kh   = (unsigned short*)ws;          ws += n_elems * sizeof(unsigned short);
  float* rowmax        = (float*)ws;                   ws += (size_t)BATCH * SEQL * sizeof(float);
  float* M             = (float*)ws;                   ws += (size_t)BATCH * SEQL * sizeof(float);
  float* kmean         = (float*)ws;                   ws += (size_t)BATCH * DIM * sizeof(float);
  float* vmean         = (float*)ws;                   ws += (size_t)BATCH * DIM * sizeof(float);
  int*   topk          = (int*)ws;

  // 1) K -> bf16
  convert_bf16_kernel<<<(int)(n_elems / 256), 256, 0, stream>>>(K, Kh, (int)n_elems);
  // 2) K_mean, V_mean
  means_kernel<<<2 * BATCH, 256, 0, stream>>>(K, V, kmean, vmean);
  // 3) dense rowmax via WMMA (8 waves/block, async triple-buffered K in LDS)
  rowmax_wmma_kernel<<<BATCH * (SEQL / 128), 256, 0, stream>>>(Q, Kh, rowmax);
  // 4) sparsity measure M
  compute_m_kernel<<<BATCH * SEQL / 4, 128, 0, stream>>>(Q, kmean, rowmax, M);
  // 5) top-k per batch
  topk_kernel<<<BATCH, 256, 0, stream>>>(M, topk);
  // 6) baseline fill with V_mean
  fill_mean_kernel<<<(int)(n_elems / 256), 256, 0, stream>>>(vmean, out);
  // 7) sparse attention overwrite for selected rows
  dim3 sgrid(KTOP, BATCH);
  sparse_attn_kernel<<<sgrid, 256, 0, stream>>>(Q, K, V, topk, out);
}
